// HyperUnet_29265907155585
// MI455X (gfx1250) — matrix operand order repaired and checked by reading, not compiled, not run on hardware
//
#include <hip/hip_runtime.h>
#include <hip/hip_bf16.h>
#include <cmath>

// ---------------------------------------------------------------------------
// HyperUNet on gfx1250 (MI455X).  Convs run as implicit GEMM on
// v_wmma_f32_16x16x32_f16 (A = per-sample generated weights, pre-laid in
// WMMA A-fragment order; B = activation patch in channel-last LDS).
//
// All activations are NHWC f16, so:
//  - LDS staging = aligned 16B vector copies straight into the channel-last
//    patch layout (no transpose),
//  - conv epilogue = packed 16B stores (8 consecutive channels per store),
//  - pool / bilinear / head are channel-contiguous.
//
// K-dimension permutation (our choice, A and B agree):
//   k = (ky*3 + kx)*IC + ic
// so each 32-wide K chunk is contiguous-in-ic -> the B fragment is a single
// 32-byte contiguous LDS load per lane (two aligned ds_load_b128).
//
// Input flattening assumption (jax pytree order, dict keys sorted):
//   d_in[0] = x (B,H,W,2) f32
//   d_in[1] = hyperparams (B,2) f32
//   d_in[2 + 4*i + {0,1,2,3}] = (Wk, bk, Wb, bb) for conv i in SORTED name
//     order: d1a,d1b,d2a,d2b,d3a,d3b,d4a,d4b,last,u1a,u1b,u2a,u2b,u3a,u3b
//   d_in[62 + 2*l + {0,1}] = (Wl, bl) for hypernet layer l = 0..3
// ---------------------------------------------------------------------------

typedef __attribute__((ext_vector_type(16))) _Float16 v16h;
typedef __attribute__((ext_vector_type(8)))  _Float16 v8h;
typedef __attribute__((ext_vector_type(8)))  float    v8f;

#define BATCH 4
#define HDIM  32
#define HCH   64

// ------------------------------ hypernetwork -------------------------------
// h = hyperparams / sum; 4 layers (2->32->32->32->32), leaky_relu(0.01).
__global__ __launch_bounds__(128)
void hnet_kernel(const float* __restrict__ hyp,
                 const float* __restrict__ W0, const float* __restrict__ b0,
                 const float* __restrict__ W1, const float* __restrict__ b1,
                 const float* __restrict__ W2, const float* __restrict__ b2,
                 const float* __restrict__ W3, const float* __restrict__ b3,
                 float* __restrict__ hout)
{
    __shared__ float cur[BATCH * HDIM];
    const int t = threadIdx.x;
    const int b = t >> 5;
    const int r = t & 31;
    const float s = hyp[b * 2 + 0] + hyp[b * 2 + 1];
    if (r < 2) cur[b * HDIM + r] = hyp[b * 2 + r] / s;
    __syncthreads();
    const float* Ws[4] = {W0, W1, W2, W3};
    const float* bs[4] = {b0, b1, b2, b3};
#pragma unroll
    for (int l = 0; l < 4; ++l) {
        const int ind = (l == 0) ? 2 : HDIM;
        float acc = bs[l][r];
        for (int d = 0; d < ind; ++d) acc += Ws[l][r * ind + d] * cur[b * HDIM + d];
        acc = acc > 0.f ? acc : 0.01f * acc;
        __syncthreads();
        cur[b * HDIM + r] = acc;
        __syncthreads();
    }
    hout[t] = cur[t];
}

// ------------------------- generated-weight fragments ----------------------
// Write conv weights directly in WMMA A-fragment order (contiguous K order used here, not AMD's table:
// lanes 0-15 hold K 0..15; lanes 16-31 hold K 16..31, matching B) with the
// permuted K order k = (ky*3+kx)*IC + ic.
// Linear half index: b*nch*2048 + chunk*2048 + octile*512 + lane*16 + j.
__global__ void gen_wfrag(const float* __restrict__ Wk, const float* __restrict__ bk,
                          const float* __restrict__ h, _Float16* __restrict__ out,
                          int IC, int nch, int total)
{
    const int t = blockIdx.x * 256 + threadIdx.x;
    if (t >= total) return;
    const int j      = t & 15;
    const int lane   = (t >> 4) & 31;
    const int octile = (t >> 9) & 3;
    const int chunk  = (t >> 11) % nch;
    const int b      = t / (nch << 11);
    const int hi     = lane >> 4;
    const int krel   = hi * 16 + j;
    const int k      = chunk * 32 + krel;
    const int oc     = octile * 16 + (lane & 15);
    const int K      = IC * 9;
    float acc = 0.f;
    if (k < K) {
        const int s   = k / IC;          // ky*3+kx
        const int icg = k - s * IC;      // input channel
        const long e  = ((long)oc * IC + icg) * 9 + s;   // original (oc,ic,ky,kx) flat
        acc = bk[e];
        const float* wr = Wk + e * HDIM;
        const float* hb = h + b * HDIM;
#pragma unroll
        for (int d = 0; d < HDIM; ++d) acc += hb[d] * wr[d];
    }
    out[t] = (_Float16)acc;
}

// out[b*N+n] = bias[n] + dot(h[b], M[n,:])  (conv biases, head weights/bias)
__global__ void gen_linmap(const float* __restrict__ M, const float* __restrict__ bias,
                           const float* __restrict__ h, float* __restrict__ out, int N)
{
    const int t = blockIdx.x * 64 + threadIdx.x;
    if (t >= BATCH * N) return;
    const int b = t / N, n = t % N;
    float acc = bias[n];
    const float* r  = M + (long)n * HDIM;
    const float* hb = h + b * HDIM;
#pragma unroll
    for (int d = 0; d < HDIM; ++d) acc += hb[d] * r[d];
    out[t] = acc;
}

// --------------------------- WMMA implicit-GEMM conv -----------------------
// Block: 256 threads = 8 waves.  Tile: 8 rows x 16 cols of output pixels,
// all 64 output channels.  Wave w owns output row w (16 pixels), keeps
// 4 accumulators (4 oc-tiles of 16).  LDS patch is channel-last:
// XLDS[(dy*18+dx)*ICP + c], ICP = IC+8 (multiple of 8 halves -> 16B aligned
// vector loads; stride 144B/272B -> conflict-light/-free).
// src0/src1/dst are NHWC f16.
template <int IC, int ICSPLIT>
__global__ __launch_bounds__(256)
void conv3x3_wmma(const _Float16* __restrict__ src0,
                  const _Float16* __restrict__ src1,
                  const _Float16* __restrict__ wfrag,
                  const float* __restrict__ bias,
                  _Float16* __restrict__ dst,
                  int H, int W)
{
    constexpr int K   = IC * 9;
    constexpr int NCH = (K + 31) / 32;
    constexpr int ICP = IC + 8;
    __shared__ __align__(16) _Float16 XLDS[180 * ICP];

    const int tid  = threadIdx.x;
    const int wave = tid >> 5;
    const int lane = tid & 31;
    const int tilesX = W >> 4;
    const int tilesY = H >> 3;
    int blk = blockIdx.x;
    const int tx = blk % tilesX;  blk /= tilesX;
    const int ty = blk % tilesY;  blk /= tilesY;
    const int b  = blk;
    const int y0 = ty * 8, x0 = tx * 16;

    // cooperative stage: NHWC global -> channel-last LDS patch (pad=1, zero OOB)
    if constexpr (IC >= 8) {
        constexpr int SEG = IC / 8;          // v8h segments per position
        for (int i = tid; i < 180 * SEG; i += 256) {
            const int pos = i / SEG;
            const int seg = i - pos * SEG;
            const int dy = pos / 18, dx = pos % 18;
            const int gy = y0 - 1 + dy, gx = x0 - 1 + dx;
            const int ch8 = seg * 8;
            v8h v = v8h{};
            if ((unsigned)gy < (unsigned)H && (unsigned)gx < (unsigned)W) {
                if (ch8 < ICSPLIT)
                    v = *reinterpret_cast<const v8h*>(
                        &src0[(((long)(b * H + gy)) * W + gx) * ICSPLIT + ch8]);
                else
                    v = *reinterpret_cast<const v8h*>(
                        &src1[(((long)(b * H + gy)) * W + gx) * (IC - ICSPLIT) + (ch8 - ICSPLIT)]);
            }
            *reinterpret_cast<v8h*>(&XLDS[pos * ICP + ch8]) = v;
        }
    } else {
        for (int i = tid; i < 180 * IC; i += 256) {
            const int pos = i / IC;
            const int c   = i - pos * IC;
            const int dy = pos / 18, dx = pos % 18;
            const int gy = y0 - 1 + dy, gx = x0 - 1 + dx;
            _Float16 v = (_Float16)0.f;
            if ((unsigned)gy < (unsigned)H && (unsigned)gx < (unsigned)W)
                v = src0[(((long)(b * H + gy)) * W + gx) * IC + c];
            XLDS[pos * ICP + c] = v;
        }
    }
    __syncthreads();

    const int col   = lane & 15;          // output x within tile = B column
    const int khalf = (lane >> 4) * 16;   // B layout: lanes 16-31 hold K 16..31
    const v16h* wfb = reinterpret_cast<const v16h*>(wfrag) + (long)b * NCH * 4 * 32;

    __builtin_prefetch(&wfb[lane], 0, 1);   // warm A fragments (global_prefetch_b8)

    v8f acc[4] = {v8f{}, v8f{}, v8f{}, v8f{}};

    if constexpr (IC >= 32) {
        constexpr int CPS = IC / 32;      // chunks per (ky,kx) slab
#pragma unroll
        for (int s = 0; s < 9; ++s) {
            const int dy = s / 3, dx = s % 3;
            const int off0 = ((wave + dy) * 18 + (col + dx)) * ICP + khalf;
#pragma unroll
            for (int cc = 0; cc < CPS; ++cc) {
                union { v16h v; v8h h2[2]; } bf;
                bf.h2[0] = *reinterpret_cast<const v8h*>(&XLDS[off0 + cc * 32]);
                bf.h2[1] = *reinterpret_cast<const v8h*>(&XLDS[off0 + cc * 32 + 8]);
                const int ch = s * CPS + cc;
#pragma unroll
                for (int t = 0; t < 4; ++t) {
                    const v16h a = wfb[(long)(ch * 4 + t) * 32 + lane];
                    acc[t] = __builtin_amdgcn_wmma_f32_16x16x32_f16(
                        false, a, false, bf.v, (short)0, acc[t], false, false);
                }
            }
        }
    } else {
        // first layer: IC=2, K=18, single chunk -> scalar gather
        union { v16h v; _Float16 e[16]; } bf;
#pragma unroll
        for (int j = 0; j < 16; ++j) {
            const int k = khalf + j;
            _Float16 v = (_Float16)0.f;
            if (k < K) {
                const int s  = k / IC;
                const int ic = k - s * IC;
                const int dy = s / 3, dx = s % 3;
                v = XLDS[((wave + dy) * 18 + (col + dx)) * ICP + ic];
            }
            bf.e[j] = v;
        }
#pragma unroll
        for (int t = 0; t < 4; ++t) {
            const v16h a = wfb[(long)t * 32 + lane];
            acc[t] = __builtin_amdgcn_wmma_f32_16x16x32_f16(
                false, a, false, bf.v, (short)0, acc[t], false, false);
        }
    }

    // epilogue: bias + ReLU, packed 16B NHWC stores.
    // D layout: VGPR i -> M = m0 + i (consecutive oc), N = lane&15 (pixel).
    const int oy = y0 + wave;
    const int m0 = (lane >> 4) * 8;
    const long obase = (((long)(b * H + oy)) * W + x0 + col) * HCH + m0;
#pragma unroll
    for (int t = 0; t < 4; ++t) {
        union { v8h v; _Float16 e[8]; } o;
#pragma unroll
        for (int i = 0; i < 8; ++i) {
            const int oc = t * 16 + m0 + i;
            float v = acc[t][i] + bias[b * HCH + oc];
            o.e[i] = (_Float16)(v > 0.f ? v : 0.f);
        }
        *reinterpret_cast<v8h*>(&dst[obase + t * 16]) = o.v;
    }
}

// ------------------------------- data movers -------------------------------
// input (B,H,W,2) f32 -> same layout f16
__global__ void cvt_f32_to_f16(const float* __restrict__ x, _Float16* __restrict__ out, int n)
{
    const int t = blockIdx.x * 256 + threadIdx.x;
    if (t < n) out[t] = (_Float16)x[t];
}

// NHWC maxpool 2x2
__global__ void maxpool2(const _Float16* __restrict__ src, _Float16* __restrict__ dst,
                         int Ho, int Wo)
{
    const int t = blockIdx.x * 256 + threadIdx.x;
    if (t >= BATCH * HCH * Ho * Wo) return;
    const int c = t % HCH;
    const int x = (t / HCH) % Wo;
    const int y = (t / (HCH * Wo)) % Ho;
    const int b = t / (HCH * Wo * Ho);
    const _Float16* p = src + (((long)(b * 2 * Ho + 2 * y)) * (2 * Wo) + 2 * x) * HCH + c;
    const long rs = (long)(2 * Wo) * HCH;
    const float a  = (float)p[0],  bb = (float)p[HCH];
    const float cc = (float)p[rs], dd = (float)p[rs + HCH];
    dst[t] = (_Float16)fmaxf(fmaxf(a, bb), fmaxf(cc, dd));
}

// NHWC bilinear x2, align_corners=True
__global__ void up2_bilinear(const _Float16* __restrict__ src, _Float16* __restrict__ dst,
                             int hi, int wi)
{
    const int ho = 2 * hi, wo = 2 * wi;
    const int t = blockIdx.x * 256 + threadIdx.x;
    if (t >= BATCH * HCH * ho * wo) return;
    const int c = t % HCH;
    const int x = (t / HCH) % wo;
    const int y = (t / (HCH * wo)) % ho;
    const int b = t / (HCH * wo * ho);
    const float fy = (float)y * (float)(hi - 1) / (float)(ho - 1);
    const float fx = (float)x * (float)(wi - 1) / (float)(wo - 1);
    const int y0i = (int)floorf(fy);
    const int x0i = (int)floorf(fx);
    const int y1 = (y0i + 1 < hi) ? y0i + 1 : hi - 1;
    const int x1 = (x0i + 1 < wi) ? x0i + 1 : wi - 1;
    const float wy = fy - (float)y0i, wx = fx - (float)x0i;
    const _Float16* p = src + ((long)b * hi * wi) * HCH + c;
    const float v00 = (float)p[((long)y0i * wi + x0i) * HCH];
    const float v01 = (float)p[((long)y0i * wi + x1) * HCH];
    const float v10 = (float)p[((long)y1 * wi + x0i) * HCH];
    const float v11 = (float)p[((long)y1 * wi + x1) * HCH];
    dst[t] = (_Float16)((v00 * (1.f - wx) + v01 * wx) * (1.f - wy) +
                        (v10 * (1.f - wx) + v11 * wx) * wy);
}

// head: out = ||x||_2(channels) + bias_last + sum_oc act*w_last   (f32 NHWC)
__global__ void final_kernel(const _Float16* __restrict__ act,
                             const float* __restrict__ wlast,
                             const float* __restrict__ blast,
                             const float* __restrict__ x,
                             float* __restrict__ out)
{
    const int t = blockIdx.x * 256 + threadIdx.x;
    if (t >= BATCH * 128 * 128) return;
    const int b = t / (128 * 128);
    const _Float16* p = act + (long)t * HCH;      // 64 contiguous channels
    const float* w = wlast + b * HCH;
    float acc = blast[b];
#pragma unroll 8
    for (int oc = 0; oc < HCH; ++oc) acc += (float)p[oc] * w[oc];
    const float x0v = x[(long)t * 2], x1v = x[(long)t * 2 + 1];
    out[t] = sqrtf(x0v * x0v + x1v * x1v) + acc;
}

// --------------------------------- driver ----------------------------------
extern "C" void kernel_launch(void* const* d_in, const int* in_sizes, int n_in,
                              void* d_out, int out_size, void* d_ws, size_t ws_size,
                              hipStream_t stream)
{
    (void)in_sizes; (void)n_in; (void)out_size; (void)ws_size;
    const float* x   = (const float*)d_in[0];
    const float* hyp = (const float*)d_in[1];
    auto P = [&](int i) { return (const float*)d_in[i]; };

    // sorted conv layer order + input channel counts (idx 8 = 'last', 1x1)
    const int icTab[15] = {2, 64, 64, 64, 64, 64, 64, 64, 64, 128, 64, 128, 64, 128, 64};

    // ---- workspace bump allocator ----
    char* ws = (char*)d_ws;
    size_t off = 0;
    auto alloc = [&](size_t bytes) -> void* {
        void* p = ws + off;
        off = (off + bytes + 255) & ~(size_t)255;
        return p;
    };

    float*     hbuf  = (float*)alloc(BATCH * HDIM * sizeof(float));
    float*     wlast = (float*)alloc(BATCH * HCH * sizeof(float));
    float*     blast = (float*)alloc(BATCH * sizeof(float));
    float*     biasA[15];
    _Float16*  wfrag[15];
    int        nchA[15];
    for (int i = 0; i < 15; ++i) {
        if (i == 8) { biasA[i] = nullptr; wfrag[i] = nullptr; nchA[i] = 0; continue; }
        nchA[i]  = (icTab[i] * 9 + 31) / 32;
        biasA[i] = (float*)alloc(BATCH * HCH * sizeof(float));
        wfrag[i] = (_Float16*)alloc((size_t)BATCH * nchA[i] * 2048 * sizeof(_Float16));
    }
    _Float16* xin     = (_Float16*)alloc((size_t)BATCH * 128 * 128 * 2 * 2);
    _Float16* bufA128 = (_Float16*)alloc((size_t)BATCH * HCH * 128 * 128 * 2);
    _Float16* bufB128 = (_Float16*)alloc((size_t)BATCH * HCH * 128 * 128 * 2);
    _Float16* skip1   = (_Float16*)alloc((size_t)BATCH * HCH * 128 * 128 * 2);
    _Float16* bufA64  = (_Float16*)alloc((size_t)BATCH * HCH * 64 * 64 * 2);
    _Float16* bufB64  = (_Float16*)alloc((size_t)BATCH * HCH * 64 * 64 * 2);
    _Float16* skip2   = (_Float16*)alloc((size_t)BATCH * HCH * 64 * 64 * 2);
    _Float16* bufA32  = (_Float16*)alloc((size_t)BATCH * HCH * 32 * 32 * 2);
    _Float16* bufB32  = (_Float16*)alloc((size_t)BATCH * HCH * 32 * 32 * 2);
    _Float16* skip3   = (_Float16*)alloc((size_t)BATCH * HCH * 32 * 32 * 2);
    _Float16* bufA16  = (_Float16*)alloc((size_t)BATCH * HCH * 16 * 16 * 2);
    _Float16* bufB16  = (_Float16*)alloc((size_t)BATCH * HCH * 16 * 16 * 2);

    // ---- 1. hypernetwork ----
    hnet_kernel<<<1, 128, 0, stream>>>(hyp, P(62), P(63), P(64), P(65),
                                       P(66), P(67), P(68), P(69), hbuf);

    // ---- 2. generated weights (fragment order, permuted K) + biases ----
    for (int i = 0; i < 15; ++i) {
        const int base = 2 + 4 * i;   // (Wk, bk, Wb, bb)
        if (i == 8) {                 // 'last': 1x1, 64->1
            gen_linmap<<<(BATCH * HCH + 63) / 64, 64, 0, stream>>>(P(base + 0), P(base + 1), hbuf, wlast, HCH);
            gen_linmap<<<1, 64, 0, stream>>>(P(base + 2), P(base + 3), hbuf, blast, 1);
            continue;
        }
        const int total = BATCH * nchA[i] * 2048;
        gen_wfrag<<<(total + 255) / 256, 256, 0, stream>>>(P(base + 0), P(base + 1), hbuf,
                                                           wfrag[i], icTab[i], nchA[i], total);
        gen_linmap<<<(BATCH * HCH + 63) / 64, 64, 0, stream>>>(P(base + 2), P(base + 3), hbuf,
                                                               biasA[i], HCH);
    }

    // ---- 3. input convert (NHWC f32 -> NHWC f16) ----
    const int nIn = BATCH * 128 * 128 * 2;
    cvt_f32_to_f16<<<(nIn + 255) / 256, 256, 0, stream>>>(x, xin, nIn);

    auto cgrid = [](int H, int W) { return dim3(BATCH * (H / 8) * (W / 16)); };
    auto c64 = [&](const _Float16* s, int li, _Float16* d, int H, int W) {
        conv3x3_wmma<64, 64><<<cgrid(H, W), 256, 0, stream>>>(s, nullptr, wfrag[li], biasA[li], d, H, W);
    };
    auto c128 = [&](const _Float16* s0, const _Float16* s1, int li, _Float16* d, int H, int W) {
        conv3x3_wmma<128, 64><<<cgrid(H, W), 256, 0, stream>>>(s0, s1, wfrag[li], biasA[li], d, H, W);
    };
    auto pool = [&](const _Float16* s, _Float16* d, int Ho, int Wo) {
        maxpool2<<<(BATCH * HCH * Ho * Wo + 255) / 256, 256, 0, stream>>>(s, d, Ho, Wo);
    };
    auto up = [&](const _Float16* s, _Float16* d, int hi, int wi) {
        up2_bilinear<<<(BATCH * HCH * 4 * hi * wi + 255) / 256, 256, 0, stream>>>(s, d, hi, wi);
    };

    // ---- 4. U-Net ----
    conv3x3_wmma<2, 2><<<cgrid(128, 128), 256, 0, stream>>>(xin, nullptr, wfrag[0], biasA[0],
                                                            bufA128, 128, 128);   // d1a
    c64(bufA128, 1, skip1, 128, 128);                                             // d1b -> c1
    pool(skip1, bufA64, 64, 64);
    c64(bufA64, 2, bufB64, 64, 64);                                               // d2a
    c64(bufB64, 3, skip2, 64, 64);                                                // d2b -> c2
    pool(skip2, bufA32, 32, 32);
    c64(bufA32, 4, bufB32, 32, 32);                                               // d3a
    c64(bufB32, 5, skip3, 32, 32);                                                // d3b -> c3
    pool(skip3, bufA16, 16, 16);
    c64(bufA16, 6, bufB16, 16, 16);                                               // d4a
    c64(bufB16, 7, bufA16, 16, 16);                                               // d4b
    up(bufA16, bufA32, 16, 16);
    c128(bufA32, skip3, 13, bufB32, 32, 32);                                      // u3a
    c64(bufB32, 14, bufA32, 32, 32);                                              // u3b
    up(bufA32, bufA64, 32, 32);
    c128(bufA64, skip2, 11, bufB64, 64, 64);                                      // u2a
    c64(bufB64, 12, bufA64, 64, 64);                                              // u2b
    up(bufA64, bufA128, 64, 64);
    c128(bufA128, skip1, 9, bufB128, 128, 128);                                   // u1a
    c64(bufB128, 10, bufA128, 128, 128);                                          // u1b

    // ---- 5. head + residual ----
    final_kernel<<<(BATCH * 128 * 128 + 255) / 256, 256, 0, stream>>>(
        bufA128, wlast, blast, x, (float*)d_out);
}
